// ShiftedWindowMHSA_17325898072403
// MI455X (gfx1250) — matrix operand order repaired and checked by reading, not compiled, not run on hardware
//
#include <hip/hip_runtime.h>

// ---------------------------------------------------------------------------
// Swin shifted-window MHSA, fully fused per-window kernel for gfx1250.
// bf16 WMMA (16x16x32) + fp32 softmax; all intermediates live in WGP LDS.
// ---------------------------------------------------------------------------

typedef __attribute__((ext_vector_type(16))) __bf16 v16bf;
typedef __attribute__((ext_vector_type(8)))  float  v8f;

#define B_WIN   4096
#define NTOK    49
#define NPAD    64
#define CDIM    128
#define NHEAD   4
#define HDIM    32
#define NWMASK  64
#define QK_SCALE 0.17677669529663689f   // 1/sqrt(32)

// LDS row strides (elements): 16B-aligned rows, strideBytes/4 has gcd<=4 with
// the 64 banks (136 -> 272B, 40 -> 80B, 72 -> 144B).
#define XS_STRIDE 136   // x / y rows of 128
#define QK_STRIDE 40    // q / k rows of 32
#define VT_STRIDE 72    // v^T rows of 64
#define PS_STRIDE 72    // P rows of 64
#define QH_ELEMS (NPAD * QK_STRIDE)   // 2560
#define VH_ELEMS (HDIM * VT_STRIDE)   // 2304
#define PH_ELEMS (NPAD * PS_STRIDE)   // 4608

union frag_pun { uint4 u[2]; v16bf v; };

// A-matrix 16x32 bf16 fragment from row-major [M][K] storage.
// Lane layout (ISA 7.12.2): elements 0..7 = row[8*half + 0..7],
// elements 8..15 = row[8*half + 16..23]  -> two 16B loads per lane.
__device__ __forceinline__ v16bf afrag(const __bf16* base, int stride, int lane) {
  int m = lane & 15, half = lane >> 4;
  const __bf16* row = base + m * stride + 8 * half;
  frag_pun p;
  p.u[0] = *(const uint4*)(row);
  p.u[1] = *(const uint4*)(row + 16);
  return p.v;
}

// B-matrix 32x16 bf16 fragment from B^T stored row-major: bt[n][k].
// elements 0..7 = row[8*half + 0..7], elements 8..15 = row[8*half + 16..23]
// -> two 16B loads per lane.
__device__ __forceinline__ v16bf bfrag(const __bf16* bt, int stride, int lane) {
  int n = lane & 15, half = lane >> 4;
  const __bf16* row = bt + n * stride + 8 * half;
  frag_pun p;
  p.u[0] = *(const uint4*)(row);
  p.u[1] = *(const uint4*)(row + 16);
  return p.v;
}

__device__ __forceinline__ v8f wmma_bf16(v16bf a, v16bf b, v8f c) {
  return __builtin_amdgcn_wmma_f32_16x16x32_bf16(false, a, false, b,
                                                 (short)0, c, false, false);
}

// ---- prep kernels ----------------------------------------------------------

__global__ void convert_weights(const float* __restrict__ qkv_w,
                                const float* __restrict__ proj_w,
                                __bf16* __restrict__ wqkv,
                                __bf16* __restrict__ wproj) {
  int i = blockIdx.x * blockDim.x + threadIdx.x;
  if (i < 384 * 128) wqkv[i] = (__bf16)qkv_w[i];
  int j = i - 384 * 128;
  if (j >= 0 && j < 128 * 128) wproj[j] = (__bf16)proj_w[j];
}

// AM[w][h][r][c] = mask[w][r][c] + bias[h][r][c]; padded rows/cols -> -1e30
__global__ void build_addmask(const float* __restrict__ mask,
                              const float* __restrict__ table,
                              const int*   __restrict__ relidx,
                              float* __restrict__ AM) {
  int idx = blockIdx.x * blockDim.x + threadIdx.x;   // [64][4][64][64]
  if (idx >= NWMASK * NHEAD * NPAD * NPAD) return;
  int c = idx & 63, r = (idx >> 6) & 63, h = (idx >> 12) & 3, w = idx >> 14;
  float v = -1e30f;
  if (r < NTOK && c < NTOK)
    v = mask[(w * NTOK + r) * NTOK + c] + table[relidx[r * NTOK + c] * NHEAD + h];
  AM[idx] = v;
}

// ---- fused per-window kernel: QKV GEMM -> MHSA -> proj GEMM ----------------

__global__ __launch_bounds__(256)
void swin_fused(const float* __restrict__ x, const __bf16* __restrict__ wqkv,
                const float* __restrict__ qkv_b, const __bf16* __restrict__ wproj,
                const float* __restrict__ proj_b, const float* __restrict__ AM,
                float* __restrict__ out) {
  __shared__ __align__(16) __bf16 xs[NPAD * XS_STRIDE];     // 17408 B (also y)
  __shared__ __align__(16) __bf16 qs[NHEAD * QH_ELEMS];     // 20480 B
  __shared__ __align__(16) __bf16 ks[NHEAD * QH_ELEMS];     // 20480 B
  __shared__ __align__(16) __bf16 vts[NHEAD * VH_ELEMS];    // 18432 B
  __shared__ __align__(16) __bf16 ps[NHEAD * PH_ELEMS];     // 36864 B
  __bf16* ys = xs;   // x staging reused for attention output

  int b = blockIdx.x, tid = threadIdx.x;
  int lane = tid & 31;
  int col = lane & 15, half = lane >> 4;
  // wave id is wave-uniform: keep it in an SGPR so tile/head indexing and the
  // q/k/v dispatch stay in scalar registers and scalar branches.
  int wave = __builtin_amdgcn_readfirstlane(tid >> 5);

  // phase 1: stage x (fp32 -> bf16), rows >= 49 zeroed
  for (int i = tid; i < NPAD * (CDIM / 4); i += 256) {   // 2048 float4 chunks
    int r = i >> 5, c4 = (i & 31) << 2;
    float4 f = make_float4(0.f, 0.f, 0.f, 0.f);
    if (r < NTOK) f = ((const float4*)(x + ((size_t)b * NTOK + r) * CDIM))[c4 >> 2];
    union { __bf16 h4[4]; uint2 u; } pk;
    pk.h4[0] = (__bf16)f.x; pk.h4[1] = (__bf16)f.y;
    pk.h4[2] = (__bf16)f.z; pk.h4[3] = (__bf16)f.w;
    *(uint2*)&xs[r * XS_STRIDE + c4] = pk.u;
  }
  __syncthreads();

  // phase 2: QKV GEMM [64x128]x[128x384], nt-major so B-fragments are loaded
  // once per output column strip and reused across all 4 M-tiles.
  for (int nt = wave; nt < 24; nt += 8) {
    const __bf16* wb = wqkv + (16 * nt) * CDIM;
    v16bf bb0 = bfrag(wb +  0, CDIM, lane);
    v16bf bb1 = bfrag(wb + 32, CDIM, lane);
    v16bf bb2 = bfrag(wb + 64, CDIM, lane);
    v16bf bb3 = bfrag(wb + 96, CDIM, lane);

    int co0   = 16 * nt;                 // scalar -> uniform branches below
    int which = co0 >> 7;
    int h     = (co0 >> 5) & 3;
    int dh    = (co0 & 31) + col;
    float bias = qkv_b[co0 + col];

#pragma unroll
    for (int mt = 0; mt < 4; ++mt) {
      const __bf16* xr = xs + (16 * mt) * XS_STRIDE;
      v8f acc = {};
      acc = wmma_bf16(afrag(xr +  0, XS_STRIDE, lane), bb0, acc);
      acc = wmma_bf16(afrag(xr + 32, XS_STRIDE, lane), bb1, acc);
      acc = wmma_bf16(afrag(xr + 64, XS_STRIDE, lane), bb2, acc);
      acc = wmma_bf16(afrag(xr + 96, XS_STRIDE, lane), bb3, acc);
      if (which == 0) {
#pragma unroll
        for (int j = 0; j < 8; ++j)
          qs[h * QH_ELEMS + (16 * mt + j + 8 * half) * QK_STRIDE + dh] =
              (__bf16)((acc[j] + bias) * QK_SCALE);
      } else if (which == 1) {
#pragma unroll
        for (int j = 0; j < 8; ++j)
          ks[h * QH_ELEMS + (16 * mt + j + 8 * half) * QK_STRIDE + dh] =
              (__bf16)(acc[j] + bias);
      } else {  // v^T: 8 consecutive rows per lane -> one packed b128 store
        union { __bf16 h8[8]; uint4 u; } pk;
#pragma unroll
        for (int j = 0; j < 8; ++j) pk.h8[j] = (__bf16)(acc[j] + bias);
        *(uint4*)&vts[h * VH_ELEMS + dh * VT_STRIDE + 16 * mt + 8 * half] = pk.u;
      }
    }
  }
  __syncthreads();

  // phase 3: attention; 2 waves per head, 32 rows each (2 M-tiles)
  {
    int h = wave >> 1, sub = wave & 1;                  // scalar
    const __bf16* qh = qs  + h * QH_ELEMS;
    const __bf16* kh = ks  + h * QH_ELEMS;
    const __bf16* vh = vts + h * VH_ELEMS;
    __bf16* ph = ps + h * PH_ELEMS;
    const float* am = AM + (size_t)((b & 63) * NHEAD + h) * NPAD * NPAD;
#pragma unroll
    for (int mt2 = 0; mt2 < 2; ++mt2) {
      int r0 = 32 * sub + 16 * mt2;
      // S = q @ k^T (Dh=32 -> single K-step per 16x16 tile)
      v16bf aq = afrag(qh + r0 * QK_STRIDE, QK_STRIDE, lane);
      v8f s[4];
#pragma unroll
      for (int t = 0; t < 4; ++t) {
        v8f z = {};
        s[t] = wmma_bf16(aq, bfrag(kh + (16 * t) * QK_STRIDE, QK_STRIDE, lane), z);
      }
#pragma unroll
      for (int t = 0; t < 4; ++t)
#pragma unroll
        for (int j = 0; j < 8; ++j)
          s[t][j] += am[(r0 + j + 8 * half) * NPAD + 16 * t + col];
      // fp32 softmax; row lives in one VGPR across 16 lanes of a half-wave
#pragma unroll
      for (int j = 0; j < 8; ++j) {
        float m = fmaxf(fmaxf(s[0][j], s[1][j]), fmaxf(s[2][j], s[3][j]));
        m = fmaxf(m, __shfl_xor(m, 1, 32));
        m = fmaxf(m, __shfl_xor(m, 2, 32));
        m = fmaxf(m, __shfl_xor(m, 4, 32));
        m = fmaxf(m, __shfl_xor(m, 8, 32));
        float e0 = __expf(s[0][j] - m), e1 = __expf(s[1][j] - m);
        float e2 = __expf(s[2][j] - m), e3 = __expf(s[3][j] - m);
        float sum = e0 + e1 + e2 + e3;
        sum += __shfl_xor(sum, 1, 32);
        sum += __shfl_xor(sum, 2, 32);
        sum += __shfl_xor(sum, 4, 32);
        sum += __shfl_xor(sum, 8, 32);
        float inv = __builtin_amdgcn_rcpf(sum);
        s[0][j] = e0 * inv; s[1][j] = e1 * inv;
        s[2][j] = e2 * inv; s[3][j] = e3 * inv;
      }
      // P -> LDS (A-fragment re-layout); only this wave's rows are touched
#pragma unroll
      for (int t = 0; t < 4; ++t)
#pragma unroll
        for (int j = 0; j < 8; ++j)
          ph[(r0 + j + 8 * half) * PS_STRIDE + 16 * t + col] = (__bf16)s[t][j];
      // LDS is in-order per wave; fence compiler + wait for the stores
      asm volatile("s_wait_dscnt 0" ::: "memory");
      v16bf pa0 = afrag(ph + r0 * PS_STRIDE,      PS_STRIDE, lane);
      v16bf pa1 = afrag(ph + r0 * PS_STRIDE + 32, PS_STRIDE, lane);
#pragma unroll
      for (int t = 0; t < 2; ++t) {
        v8f acc = {};
        acc = wmma_bf16(pa0, bfrag(vh + (16 * t) * VT_STRIDE,      VT_STRIDE, lane), acc);
        acc = wmma_bf16(pa1, bfrag(vh + (16 * t) * VT_STRIDE + 32, VT_STRIDE, lane), acc);
#pragma unroll
        for (int j = 0; j < 8; ++j)
          ys[(r0 + j + 8 * half) * XS_STRIDE + h * HDIM + 16 * t + col] =
              (__bf16)acc[j];
      }
    }
  }
  __syncthreads();

  // phase 4: output projection; each wave owns one 16-wide output column strip
  {
    int nt = wave;                                      // scalar, 0..7
    const __bf16* wb = wproj + (16 * nt) * CDIM;
    v16bf bb0 = bfrag(wb +  0, CDIM, lane);
    v16bf bb1 = bfrag(wb + 32, CDIM, lane);
    v16bf bb2 = bfrag(wb + 64, CDIM, lane);
    v16bf bb3 = bfrag(wb + 96, CDIM, lane);
    float bias = proj_b[16 * nt + col];
#pragma unroll
    for (int mt = 0; mt < 4; ++mt) {
      const __bf16* yr = ys + (16 * mt) * XS_STRIDE;
      v8f acc = {};
      acc = wmma_bf16(afrag(yr +  0, XS_STRIDE, lane), bb0, acc);
      acc = wmma_bf16(afrag(yr + 32, XS_STRIDE, lane), bb1, acc);
      acc = wmma_bf16(afrag(yr + 64, XS_STRIDE, lane), bb2, acc);
      acc = wmma_bf16(afrag(yr + 96, XS_STRIDE, lane), bb3, acc);
#pragma unroll
      for (int j = 0; j < 8; ++j) {
        int r = 16 * mt + j + 8 * half;
        if (r < NTOK)
          out[((size_t)b * NTOK + r) * CDIM + 16 * nt + col] = acc[j] + bias;
      }
    }
  }
}

// ---------------------------------------------------------------------------

extern "C" void kernel_launch(void* const* d_in, const int* in_sizes, int n_in,
                              void* d_out, int out_size, void* d_ws, size_t ws_size,
                              hipStream_t stream) {
  const float* x      = (const float*)d_in[0];
  const float* mask   = (const float*)d_in[1];
  const float* qkv_w  = (const float*)d_in[2];
  const float* qkv_b  = (const float*)d_in[3];
  const float* proj_w = (const float*)d_in[4];
  const float* proj_b = (const float*)d_in[5];
  const float* rbtab  = (const float*)d_in[6];
  const int*   relidx = (const int*)d_in[7];

  char* ws = (char*)d_ws;
  const size_t OFF_WQKV  = 0;            // 384*128*2 = 98304
  const size_t OFF_WPROJ = 98304;        // 128*128*2 = 32768
  const size_t OFF_AM    = 131072;       // 64*4*64*64*4 = 4 MiB
  __bf16* wqkv  = (__bf16*)(ws + OFF_WQKV);
  __bf16* wproj = (__bf16*)(ws + OFF_WPROJ);
  float*  AM    = (float*)(ws + OFF_AM);

  convert_weights<<<(384 * 128 + 128 * 128 + 255) / 256, 256, 0, stream>>>(
      qkv_w, proj_w, wqkv, wproj);
  build_addmask<<<(NWMASK * NHEAD * NPAD * NPAD) / 256, 256, 0, stream>>>(
      mask, rbtab, relidx, AM);
  swin_fused<<<B_WIN, 256, 0, stream>>>(x, wqkv, qkv_b, wproj, proj_b, AM,
                                        (float*)d_out);
}
